// Selfattention_21904333209638
// MI455X (gfx1250) — matrix operand order repaired and checked by reading, not compiled, not run on hardware
//
#include <hip/hip_runtime.h>
#include <hip/hip_bf16.h>

typedef __attribute__((ext_vector_type(16))) _Float16 v16h;
typedef __attribute__((ext_vector_type(8)))  float    v8f;
typedef __attribute__((ext_vector_type(4)))  unsigned int v4u;
typedef __attribute__((ext_vector_type(8)))  int      v8i;
typedef __attribute__((ext_vector_type(4)))  int      v4i;

#define B_ 4
#define S_ 4096
#define E_ 256
#define U_ 128
#define ROWS 64
#define NEG_FILL_F (-4294967295.0f)   // -2^32 + 1

// per-32-key chunk in workspace: kT frags (8KB) + V frags (16KB) + mask (128B)
#define CHUNK_B   24704
#define V_HOFF    4096                 // half-offset of V frags inside chunk
#define MASK_BOFF 24576                // byte-offset of mask inside chunk

__device__ __forceinline__ float grp16_max(float v) {
#pragma unroll
  for (int m = 8; m >= 1; m >>= 1) v = fmaxf(v, __shfl_xor(v, m, 32));
  return v;
}
__device__ __forceinline__ float grp16_sum(float v) {
#pragma unroll
  for (int m = 8; m >= 1; m >>= 1) v += __shfl_xor(v, m, 32);
  return v;
}

// ---------------------------------------------------------------------------
// TDM: 1-D contiguous copy global -> LDS, 8-byte elements (ISA 08 Sec.8 D#).
// ---------------------------------------------------------------------------
__device__ __forceinline__ void tdm_load_1d(unsigned lds_addr, const void* gsrc,
                                            unsigned nbytes) {
  const unsigned n8 = nbytes >> 3;                       // 8-byte elements
  const unsigned long long ga = (unsigned long long)(uintptr_t)gsrc;
  v4u g0;
  g0.x = 1u;                                             // count=1 (valid, load)
  g0.y = lds_addr;                                       // lds_addr[31:0]
  g0.z = (unsigned)(ga & 0xffffffffull);                 // global_addr[31:0]
  g0.w = ((unsigned)(ga >> 32) & 0x01ffffffu) | (2u << 30); // ga[56:32]|type=2
  v8i g1;
  g1[0] = (int)(3u << 16);                               // data_size=3 (8B)
  g1[1] = (int)((n8 & 0xffffu) << 16);                   // tensor_dim0[15:0]
  g1[2] = (int)(((n8 >> 16) & 0xffffu) | (1u << 16));    // td0[31:16] | td1=1
  g1[3] = (int)((n8 & 0xffffu) << 16);                   // tile_dim0 = n8
  g1[4] = 0;
  g1[5] = (int)n8;                                       // tensor_dim0_stride
  g1[6] = 0;
  g1[7] = 0;
  v4i gz = {0, 0, 0, 0};
#if defined(__clang_major__) && __clang_major__ >= 23
  v8i gz8 = {0, 0, 0, 0, 0, 0, 0, 0};
  __builtin_amdgcn_tensor_load_to_lds(g0, g1, gz, gz, gz8, 0);
#else
  __builtin_amdgcn_tensor_load_to_lds(g0, g1, gz, gz, 0);
#endif
}

// ---------------------------------------------------------------------------
// Kernel 0: convert weights f32->f16 pre-swizzled into WMMA B-fragment order.
// frag index: q: ct*8+chunk (64), k: 64+..., v: 128+ct*8+chunk (128). Each
// fragment = 32 lanes x 16 halves; element j of lane = W[kc+16*(j>>3)+8*(lane>>4)+(j&7)][colb+lane&15].
// ---------------------------------------------------------------------------
__global__ __launch_bounds__(256) void weight_swizzle_kernel(
    const float* __restrict__ Wq, const float* __restrict__ Wk,
    const float* __restrict__ Wv, _Float16* __restrict__ wfrag)
{
  const int t = blockIdx.x * 256 + threadIdx.x;          // 0..8191
  const int fi = t >> 5, lane = t & 31;
  const int half = lane >> 4, ln = lane & 15;
  const float* W; int ldw, colb;
  if (fi < 64)       { W = Wq; ldw = U_; colb = (fi >> 3) * 16; }
  else if (fi < 128) { W = Wk; ldw = U_; colb = ((fi - 64) >> 3) * 16; }
  else               { W = Wv; ldw = E_; colb = ((fi - 128) >> 3) * 16; }
  const int chunk = fi & 7;
  v16h frag;
#pragma unroll
  for (int j = 0; j < 16; ++j) {
    const int K = chunk * 32 + ((j >> 3) << 4) + 8 * half + (j & 7);
    frag[j] = (_Float16)W[(size_t)K * ldw + colb + ln];
  }
  *(v16h*)(wfrag + (size_t)t * 16) = frag;
}

// ---------------------------------------------------------------------------
// Kernel 1: QKV projection with WMMA; outputs pre-swizzled for kernel 2:
//   q -> A-fragment order [b][s/16][chunk][lane][16]
//   k,v,mask -> contiguous per-32-key chunks in B-fragment order
// ---------------------------------------------------------------------------
__global__ __launch_bounds__(256) void qkv_proj_kernel(
    const float* __restrict__ x,
    const float* __restrict__ bq, const float* __restrict__ bk,
    const float* __restrict__ bv, const _Float16* __restrict__ wfrag,
    _Float16* __restrict__ q_frag, char* __restrict__ kv_ws)
{
  __shared__ __align__(32) _Float16 xs_frag[4][8][32][16];   // 32 KB, A-frag order
  const int tid = threadIdx.x;
  const int blocksPerB = S_ / ROWS;
  const int b  = blockIdx.x / blocksPerB;
  const int s0 = (blockIdx.x % blocksPerB) * ROWS;
  const float* xb = x + ((size_t)b * S_ + s0) * E_;

  // stage x into A-fragment layout (f32 -> f16)
  for (int i = tid; i < ROWS * E_; i += 256) {
    const int s_l = i >> 8, e = i & 255;
    const int rt = s_l >> 4, chunk = e >> 5, ul = e & 31;
    const int j = ((ul >> 4) << 3) | (ul & 7);
    const int lane_a = (s_l & 15) | (((ul >> 3) & 1) << 4);
    xs_frag[rt][chunk][lane_a][j] = (_Float16)xb[i];
  }
  // padding mask: sign(sum |x_row|), written into the kv chunk
  if (tid < ROWS) {
    float s = 0.f;
    const float* row = xb + (size_t)tid * E_;
    for (int j = 0; j < E_; ++j) s += fabsf(row[j]);
    const int srow = s0 + tid;
    char* chunkp = kv_ws + ((size_t)b * (S_ / 32) + (srow >> 5)) * CHUNK_B;
    *(float*)(chunkp + MASK_BOFF + (size_t)(srow & 31) * 4) = (s > 0.f) ? 1.f : 0.f;
  }
  __syncthreads();

  const int lane = tid & 31, wv = tid >> 5;
  const int half = lane >> 4, ln = lane & 15;

  for (int tile = wv; tile < 128; tile += 8) {
    const int rt = tile >> 5;
    const int c  = tile & 31;
    const float* bias; int colb, kind, fbase;
    if (c < 8)       { bias = bq; colb = c * 16;        kind = 0; fbase = c * 8; }
    else if (c < 16) { bias = bk; colb = (c - 8) * 16;  kind = 1; fbase = 64 + (c - 8) * 8; }
    else             { bias = bv; colb = (c - 16) * 16; kind = 2; fbase = 128 + (c - 16) * 8; }

    v8f acc = {};
#pragma unroll
    for (int chunk = 0; chunk < 8; ++chunk) {
      const v16h a  = *(const v16h*)&xs_frag[rt][chunk][lane][0];
      const v16h bf = *(const v16h*)(wfrag + ((size_t)(fbase + chunk) * 32 + lane) * 16);
      acc = __builtin_amdgcn_wmma_f32_16x16x32_f16(false, a, false, bf,
                                                   (short)0, acc, false, false);
    }
    const int col = colb + ln;
    const float bb = bias[col];
#pragma unroll
    for (int vv = 0; vv < 8; ++vv) {
      float z = acc[vv] + bb;
      z = (z >= 0.f) ? z : 0.2f * z;                    // leaky_relu(0.2)
      const _Float16 h = (_Float16)z;
      const int srow = s0 + rt * 16 + vv + 8 * half;    // C-layout row
      if (kind == 0) {
        // q: A-fragment order
        const int ul = col & 31;
        const int j = ((ul >> 4) << 3) | (ul & 7);
        const int lane_a = (srow & 15) | (((ul >> 3) & 1) << 4);
        q_frag[((((size_t)b * (S_ / 16) + (srow >> 4)) * 4 + (col >> 5)) * 32 + lane_a) * 16 + j] = h;
      } else {
        _Float16* chunkp = (_Float16*)(kv_ws +
            ((size_t)b * (S_ / 32) + (srow >> 5)) * CHUNK_B);
        const int kk = srow & 31;
        if (kind == 1) {
          // kT: B-frag (K-dim = u), frag = chunk_u*2 + keytile
          const int hb = ((col & 31) >> 3) & 1, jb = (((col & 31) >> 4) << 3) | (col & 7);
          const int kt = kk >> 4, n = kk & 15;
          chunkp[((((col >> 5) * 2 + kt) * 32) + (hb * 16 + n)) * 16 + jb] = h;
        } else {
          // v: B-frag (K-dim = key), frag = col-tile
          const int hb = (kk >> 3) & 1, jb = ((kk >> 4) << 3) | (kk & 7);
          chunkp[V_HOFF + (((col >> 4) * 32) + (hb * 16 + (col & 15))) * 16 + jb] = h;
        }
      }
    }
  }
}

// ---------------------------------------------------------------------------
// Kernel 2: flash attention (TDM double-buffered K/V tiles) + residual + LN.
// One block = 64 query rows; wave w: row-tile rt=w&3, col-half ch=w>>2.
// ---------------------------------------------------------------------------
__global__ __launch_bounds__(256) void attn_kernel(
    const float* __restrict__ x,
    const _Float16* __restrict__ q_frag, const char* __restrict__ kv_ws,
    const float* __restrict__ gamma, const float* __restrict__ beta,
    float* __restrict__ out)
{
  __shared__ __align__(32) char tiles[2][CHUNK_B];        // 48.2 KB double buffer
  __shared__ __align__(32) _Float16 p_s[8][32][16];       // per-wave P (A-frag order)
  __shared__ float red_sum[4][16][2];
  __shared__ float red_sq[4][16][2];

  const int tid = threadIdx.x;
  const int blocksPerB = S_ / ROWS;
  const int b  = blockIdx.x / blocksPerB;
  const int q0 = (blockIdx.x % blocksPerB) * ROWS;
  const int lane = tid & 31, wv = tid >> 5;
  const int half = lane >> 4, ln = lane & 15;
  const int rt = wv & 3, ch = wv >> 2;

  // Q A-fragments (16 rows x 128 u), resident in registers
  v16h qa[4];
  const _Float16* qf = q_frag +
      (((size_t)b * (S_ / 16) + (q0 >> 4) + rt) * 4) * 512;
#pragma unroll
  for (int chunk = 0; chunk < 4; ++chunk)
    qa[chunk] = *(const v16h*)(qf + (chunk * 32 + lane) * 16);

  v8f acc[8];
  v8f zero = {};
#pragma unroll
  for (int t = 0; t < 8; ++t) acc[t] = zero;
  float m_run[8], l_run[8];
#pragma unroll
  for (int v = 0; v < 8; ++v) { m_run[v] = -1e30f; l_run[v] = 0.f; }

  const float scale = 0.08838834764831845f;   // 1/sqrt(128)
  const char* kvbase = kv_ws + (size_t)b * (S_ / 32) * CHUNK_B;
  const int niter = S_ / 32;

  // preload first K/V chunk via Tensor Data Mover
  if (wv == 0) {
    tdm_load_1d((unsigned)(uintptr_t)(void*)tiles[0], kvbase, CHUNK_B);
    __builtin_amdgcn_s_wait_tensorcnt(0);
  }
  __syncthreads();

  for (int it = 0; it < niter; ++it) {
    const int buf = it & 1;
    if (wv == 0 && it + 1 < niter)
      tdm_load_1d((unsigned)(uintptr_t)(void*)tiles[buf ^ 1],
                  kvbase + (size_t)(it + 1) * CHUNK_B, CHUNK_B);

    const _Float16* kT_f = (const _Float16*)tiles[buf];
    const _Float16* v_f  = (const _Float16*)(tiles[buf] + V_HOFF * 2);
    const float*    mkf  = (const float*)(tiles[buf] + MASK_BOFF);

    // S = q . kT : one 16x32 block per wave (8 WMMAs)
    v8f s0 = {}, s1 = {};
#pragma unroll
    for (int chunk = 0; chunk < 4; ++chunk) {
      const v16h b0 = *(const v16h*)(kT_f + ((chunk * 2 + 0) * 32 + lane) * 16);
      const v16h b1 = *(const v16h*)(kT_f + ((chunk * 2 + 1) * 32 + lane) * 16);
      s0 = __builtin_amdgcn_wmma_f32_16x16x32_f16(false, qa[chunk], false, b0,
                                                  (short)0, s0, false, false);
      s1 = __builtin_amdgcn_wmma_f32_16x16x32_f16(false, qa[chunk], false, b1,
                                                  (short)0, s1, false, false);
    }

    // scale + mask + online softmax; write P straight into A-fragment order
    const float mk0 = mkf[ln], mk1 = mkf[16 + ln];
    const int la = ((ln >> 3) << 4);         // A-frag half select from column
    const int j0 = ln & 7, j1 = 8 + (ln & 7);
#pragma unroll
    for (int v = 0; v < 8; ++v) {
      const float a0 = (mk0 == 0.f) ? NEG_FILL_F : s0[v] * scale;
      const float a1 = (mk1 == 0.f) ? NEG_FILL_F : s1[v] * scale;
      const float rm = grp16_max(fmaxf(a0, a1));
      const float m_new = fmaxf(m_run[v], rm);
      const float alpha = __expf(m_run[v] - m_new);
      const float p0 = __expf(a0 - m_new);
      const float p1 = __expf(a1 - m_new);
      const float rs = grp16_sum(p0 + p1);
      l_run[v] = l_run[v] * alpha + rs;
      m_run[v] = m_new;
#pragma unroll
      for (int t = 0; t < 8; ++t) acc[t][v] *= alpha;
      const int row = v + 8 * half;
      p_s[wv][row + la][j0] = (_Float16)p0;
      p_s[wv][row + la][j1] = (_Float16)p1;
    }
    asm volatile("s_wait_dscnt 0" ::: "memory");     // wave-local LDS handoff

    const v16h pa = *(const v16h*)&p_s[wv][lane][0];
#pragma unroll
    for (int t = 0; t < 8; ++t) {
      const v16h bf = *(const v16h*)(v_f + ((ch * 8 + t) * 32 + lane) * 16);
      acc[t] = __builtin_amdgcn_wmma_f32_16x16x32_f16(false, pa, false, bf,
                                                      (short)0, acc[t], false, false);
    }

    if (wv == 0 && it + 1 < niter) __builtin_amdgcn_s_wait_tensorcnt(0);
    __syncthreads();                                  // publish next buffer
  }

  // Epilogue: 1/l, residual, LayerNorm (cross-wave pair via LDS)
#pragma unroll
  for (int v = 0; v < 8; ++v) {
    const float inv = 1.0f / l_run[v];
    const int row = q0 + rt * 16 + v + 8 * half;
    float ps = 0.f, pss = 0.f;
#pragma unroll
    for (int t = 0; t < 8; ++t) {
      const int col = ch * 128 + t * 16 + ln;
      const float val = acc[t][v] * inv + x[((size_t)b * S_ + row) * E_ + col];
      acc[t][v] = val;
      ps += val; pss += val * val;
    }
    ps = grp16_sum(ps); pss = grp16_sum(pss);
    if (ln == 0) {
      red_sum[rt][v + 8 * half][ch] = ps;
      red_sq[rt][v + 8 * half][ch]  = pss;
    }
  }
  __syncthreads();
#pragma unroll
  for (int v = 0; v < 8; ++v) {
    const int lr = v + 8 * half;
    const int row = q0 + rt * 16 + lr;
    const float sum = red_sum[rt][lr][0] + red_sum[rt][lr][1];
    const float sq  = red_sq[rt][lr][0] + red_sq[rt][lr][1];
    const float mean = sum * (1.0f / E_);
    const float var  = sq * (1.0f / E_) - mean * mean;
    const float rstd = rsqrtf(var + 1e-8f);
#pragma unroll
    for (int t = 0; t < 8; ++t) {
      const int col = ch * 128 + t * 16 + ln;
      const float y = (acc[t][v] - mean) * rstd * gamma[col] + beta[col];
      out[((size_t)b * S_ + row) * E_ + col] = y;
    }
  }
}

// ---------------------------------------------------------------------------
extern "C" void kernel_launch(void* const* d_in, const int* in_sizes, int n_in,
                              void* d_out, int out_size, void* d_ws, size_t ws_size,
                              hipStream_t stream) {
  const float* x     = (const float*)d_in[0];
  const float* Wq    = (const float*)d_in[1];
  const float* bq    = (const float*)d_in[2];
  const float* Wk    = (const float*)d_in[3];
  const float* bk    = (const float*)d_in[4];
  const float* Wv    = (const float*)d_in[5];
  const float* bv    = (const float*)d_in[6];
  const float* gamma = (const float*)d_in[7];
  const float* beta  = (const float*)d_in[8];
  float* out = (float*)d_out;

  char*     kv_ws  = (char*)d_ws;                                   // B*(S/32)*CHUNK_B
  _Float16* q_frag = (_Float16*)(kv_ws + (size_t)B_ * (S_ / 32) * CHUNK_B);
  _Float16* wfrag  = q_frag + (size_t)B_ * S_ * U_;                 // 256 frags

  weight_swizzle_kernel<<<32, 256, 0, stream>>>(Wq, Wk, Wv, wfrag);
  dim3 grid(B_ * (S_ / ROWS));
  qkv_proj_kernel<<<grid, 256, 0, stream>>>(x, bq, bk, bv, wfrag, q_frag, kv_ws);
  attn_kernel<<<grid, 256, 0, stream>>>(x, q_frag, kv_ws, gamma, beta, out);
}
